// MultiAttention_8718783611467
// MI455X (gfx1250) — matrix operand order repaired and checked by reading, not compiled, not run on hardware
//
#include <hip/hip_runtime.h>
#include <hip/hip_bf16.h>

typedef __attribute__((ext_vector_type(16))) _Float16 v16h;
typedef __attribute__((ext_vector_type(8)))  _Float16 h8;
typedef __attribute__((ext_vector_type(8)))  float    v8f;
typedef __attribute__((ext_vector_type(4)))  unsigned int v4u;
typedef __attribute__((ext_vector_type(8)))  int      v8i;
typedef __attribute__((ext_vector_type(4)))  int      v4i;

#define KB 4
#define KS 2048
#define KH 8
#define KD 128
#define KHD 1024   /* KH*KD */

#if __has_builtin(__builtin_amdgcn_tensor_load_to_lds)
#define HAVE_TDM 1
#else
#define HAVE_TDM 0
#endif

__device__ __forceinline__ v8f wmma_f16(v16h a, v16h b, v8f c) {
  // D = A(16x32 f16) * B(32x16 f16) + C(16x16 f32)
  return __builtin_amdgcn_wmma_f32_16x16x32_f16(false, a, false, b, (short)0, c,
                                                false, false);
}

__device__ __forceinline__ v16h pack16(h8 lo, h8 hi) {
  v16h r;
#pragma unroll
  for (int e = 0; e < 8; ++e) { r[e] = lo[e]; r[e + 8] = hi[e]; }
  return r;
}

// ---------------------------------------------------------------------------
// Tensor Data Mover: 2D tile (2-byte elements) global -> LDS.
// D# built per CDNA5 ISA ch.8: group0 = flags/lds_addr/global_addr/type,
// group1 = data_size, tensor dims, tile dims, dim0 stride. Groups 2/3 unused
// (2D tensor => both NULL-equivalent zeros).
// ---------------------------------------------------------------------------
__device__ __forceinline__ unsigned int lds_off32(const void* p) {
  // LDS aperture: LDS address is the low 32 bits of the flat address
  return (unsigned int)(unsigned long long)(uintptr_t)p;
}

#if HAVE_TDM
__device__ __forceinline__ void tdm_load_2d(unsigned int lds_addr,
                                            const void* gptr,
                                            unsigned int tensor_d0,
                                            unsigned int tensor_d1,
                                            unsigned int tile_d0,
                                            unsigned int tile_d1,
                                            unsigned int stride0) {
  unsigned long long ga = (unsigned long long)(uintptr_t)gptr;
  v4u g0;
  g0[0] = 1u;                                   // count=1, user descriptor
  g0[1] = lds_addr;                             // LDS byte address
  g0[2] = (unsigned int)(ga & 0xFFFFFFFFu);     // global_addr[31:0]
  g0[3] = (unsigned int)((ga >> 32) & 0x01FFFFFFu) | (2u << 30);  // type=2
  v8i g1;
  g1[0] = (int)(1u << 16);                                  // data_size=2B
  g1[1] = (int)((tensor_d0 & 0xFFFFu) << 16);               // t_dim0 lo16
  g1[2] = (int)(((tensor_d0 >> 16) & 0xFFFFu) |
                ((tensor_d1 & 0xFFFFu) << 16));             // t_dim0 hi/t_dim1 lo
  g1[3] = (int)(((tensor_d1 >> 16) & 0xFFFFu) |
                ((tile_d0 & 0xFFFFu) << 16));               // t_dim1 hi/tile0
  g1[4] = (int)(tile_d1 & 0xFFFFu);                         // tile1, tile2=0
  g1[5] = (int)stride0;                                     // dim0 stride lo32
  g1[6] = 0;
  g1[7] = 0;
  v4i z4 = {};
#if __clang_major__ >= 23
  v8i z8 = {};
  __builtin_amdgcn_tensor_load_to_lds(g0, g1, z4, z4, z8, 0);
#else
  __builtin_amdgcn_tensor_load_to_lds(g0, g1, z4, z4, 0);
#endif
}
#endif  // HAVE_TDM

// ---------------------------------------------------------------------------
// Kernel 1: X[8192x128] @ W[128x1024] + bias -> fp16, head-split layout.
//   vt_mode==0 : out[b][h][s][d]   (Q and K)
//   vt_mode==1 : out[b][h][d][s]   (V transposed, so PV B-frags are contiguous)
// One wave computes one 16x16 output tile; K-loop = 4 x WMMA 16x16x32.
// ---------------------------------------------------------------------------
__global__ __launch_bounds__(256) void proj_kernel(
    const float* __restrict__ X, const float* __restrict__ W,
    const float* __restrict__ bias, _Float16* __restrict__ out, float scale,
    int vt_mode) {
  const int tid = threadIdx.x;
  const int lane = tid & 31;
  const int w = tid >> 5;
  const int g = lane >> 4;
  const int col = lane & 15;

  const int tile = blockIdx.x * 8 + w;  // 512*64 tiles total
  const int nt = tile & 63;
  const int mt = tile >> 6;
  const int m0 = mt * 16;
  const int n = nt * 16 + col;

  const float* arow = X + (size_t)(m0 + col) * KD;

  v8f acc = {};
#pragma unroll
  for (int kc = 0; kc < 4; ++kc) {
    const int k0 = kc * 32;
    v16h a;
    const float* pa0 = arow + k0 + 16 * g;
    const float* pa1 = arow + k0 + 16 * g + 8;
#pragma unroll
    for (int e = 0; e < 8; ++e) {
      a[e]     = (_Float16)(pa0[e] * scale);
      a[e + 8] = (_Float16)(pa1[e] * scale);
    }
    v16h bf;
    const float* pb = W + (size_t)(k0 + 16 * g) * KHD + n;
#pragma unroll
    for (int e = 0; e < 16; ++e) bf[e] = (_Float16)pb[(size_t)e * KHD];
    acc = wmma_f16(a, bf, acc);
  }

  const float bv = bias[n];
  const int b = m0 >> 11;
  const int s0 = m0 & (KS - 1);
  const int h = n >> 7;
  const int d = n & 127;

  if (!vt_mode) {
    _Float16* o = out + ((size_t)(b * KH + h) * KS + s0) * KD + d;
#pragma unroll
    for (int v = 0; v < 8; ++v)
      o[(size_t)(v + 8 * g) * KD] = (_Float16)(acc[v] + bv);
  } else {
    h8 pk;
#pragma unroll
    for (int v = 0; v < 8; ++v) pk[v] = (_Float16)(acc[v] + bv);
    _Float16* o = out + ((size_t)(b * KH + h) * KD + d) * KS + s0 + 8 * g;
    *(h8*)o = pk;
  }
}

// ---------------------------------------------------------------------------
// Kernel 2: flash attention, TDM-staged.
// Block = 8 waves = 128 query rows of one (b,h). The 8 waves share each
// 32-key K/V^T block staged in LDS (8 KB + 8 KB), double-buffered, fetched by
// the Tensor Data Mover (wave 0 issues, TENSORcnt + barrier publishes), so
// the next block's DMA overlaps the current block's 16 WMMAs and global
// traffic drops 8x vs per-wave streaming.
// ---------------------------------------------------------------------------
__global__ __launch_bounds__(256) void flash_attn_kernel(
    const _Float16* __restrict__ Q, const _Float16* __restrict__ K,
    const _Float16* __restrict__ Vt, _Float16* __restrict__ attn) {
  __shared__ __align__(32) _Float16 Ks[2][32][KD];   // 16 KB: [key][d]
  __shared__ __align__(32) _Float16 Vts[2][KD][32];  // 16 KB: [d][key]
  __shared__ __align__(32) _Float16 Ps[8][16][32];   // 8 KB: per-wave P transpose

  const int tid = threadIdx.x;
  const int lane = tid & 31;
  const int w = tid >> 5;
  const int g = lane >> 4;
  const int col = lane & 15;

  const int bh = blockIdx.y;  // 0..31
  const int b = bh >> 3;
  const int h = bh & 7;
  const int q0 = (blockIdx.x * 8 + w) * 16;

  const _Float16* Qb = Q + (size_t)bh * KS * KD;
  const _Float16* Kb = K + (size_t)bh * KS * KD;
  const _Float16* Vb = Vt + (size_t)bh * KD * KS;

  // Q fragments (1/sqrt(d) folded in at projection), held in VGPRs throughout
  v16h qf[4];
  const _Float16* qrow = Qb + (size_t)(q0 + col) * KD;
#pragma unroll
  for (int kc = 0; kc < 4; ++kc) {
    h8 lo = *(const h8*)(qrow + kc * 32 + 16 * g);
    h8 hi = *(const h8*)(qrow + kc * 32 + 16 * g + 8);
    qf[kc] = pack16(lo, hi);
  }

  v8f acc[8];
#pragma unroll
  for (int dt = 0; dt < 8; ++dt) acc[dt] = {};
  float mi[8], li[8];
#pragma unroll
  for (int v = 0; v < 8; ++v) { mi[v] = -1e30f; li[v] = 0.0f; }

  // ---- stage key block 0 ----
#if HAVE_TDM
  if (w == 0) {
    tdm_load_2d(lds_off32(&Ks[0][0][0]), Kb, KD, KS, KD, 32, KD);
    tdm_load_2d(lds_off32(&Vts[0][0][0]), Vb, KS, KD, 32, KD, KS);
    __builtin_amdgcn_s_wait_tensorcnt(0);
  }
#else
  // cooperative fallback: 256 threads copy 8KB + 8KB
  for (int i = tid; i < 32 * KD / 8; i += 256) {
    ((h8*)&Ks[0][0][0])[i] = ((const h8*)Kb)[i];
    ((h8*)&Vts[0][0][0])[i] = *(const h8*)(Vb + (size_t)(i >> 2) * KS + (i & 3) * 8);
  }
#endif
  __syncthreads();

  for (int j = 0; j < KS / 32; ++j) {
    const int cur = j & 1;
    const int nxt = cur ^ 1;

    // ---- prefetch next key block via TDM (overlaps compute) ----
#if HAVE_TDM
    if (w == 0 && j + 1 < KS / 32) {
      const int kb2 = (j + 1) * 32;
      tdm_load_2d(lds_off32(&Ks[nxt][0][0]), Kb + (size_t)kb2 * KD,
                  KD, KS, KD, 32, KD);
      tdm_load_2d(lds_off32(&Vts[nxt][0][0]), Vb + kb2,
                  KS, KD, 32, KD, KS);
    }
#endif

    // ---- scores: two 16x16 tiles over d=128, B-frags from LDS ----
    v8f s0v = {}, s1v = {};
#pragma unroll
    for (int kc = 0; kc < 4; ++kc) {
      v16h b0 = *(const v16h*)&Ks[cur][col][kc * 32 + 16 * g];
      v16h b1 = *(const v16h*)&Ks[cur][16 + col][kc * 32 + 16 * g];
      s0v = wmma_f16(qf[kc], b0, s0v);
      s1v = wmma_f16(qf[kc], b1, s1v);
    }

    // ---- online softmax (rows v+8g live in one 16-lane half) ----
    float p0[8], p1[8], corr[8];
#pragma unroll
    for (int v = 0; v < 8; ++v) {
      float mx = fmaxf(s0v[v], s1v[v]);
      mx = fmaxf(mx, __shfl_xor(mx, 1, 32));
      mx = fmaxf(mx, __shfl_xor(mx, 2, 32));
      mx = fmaxf(mx, __shfl_xor(mx, 4, 32));
      mx = fmaxf(mx, __shfl_xor(mx, 8, 32));
      const float mnew = fmaxf(mi[v], mx);
      corr[v] = __expf(mi[v] - mnew);
      mi[v] = mnew;
      const float a0 = __expf(s0v[v] - mnew);
      const float a1 = __expf(s1v[v] - mnew);
      p0[v] = a0;
      p1[v] = a1;
      float rs = a0 + a1;
      rs += __shfl_xor(rs, 1, 32);
      rs += __shfl_xor(rs, 2, 32);
      rs += __shfl_xor(rs, 4, 32);
      rs += __shfl_xor(rs, 8, 32);
      li[v] = li[v] * corr[v] + rs;
    }
#pragma unroll
    for (int dt = 0; dt < 8; ++dt)
#pragma unroll
      for (int v = 0; v < 8; ++v) acc[dt][v] *= corr[v];

    // ---- transpose P (C-layout) -> A-layout via wave-private LDS ----
#pragma unroll
    for (int v = 0; v < 8; ++v) {
      Ps[w][v + 8 * g][col] = (_Float16)p0[v];
      Ps[w][v + 8 * g][16 + col] = (_Float16)p1[v];
    }
    v16h pf;
    {
      h8 lo = *(const h8*)&Ps[w][col][16 * g];
      h8 hi = *(const h8*)&Ps[w][col][16 * g + 8];
      pf = pack16(lo, hi);
    }

    // ---- O += P(16x32) * V(32x128): V^T rows contiguous along keys ----
#pragma unroll
    for (int dt = 0; dt < 8; ++dt) {
      v16h bv = *(const v16h*)&Vts[cur][dt * 16 + col][16 * g];
      acc[dt] = wmma_f16(pf, bv, acc[dt]);
    }

    // ---- publish next buffer / protect current one ----
#if HAVE_TDM
    if (w == 0) __builtin_amdgcn_s_wait_tensorcnt(0);
    __syncthreads();
#else
    __syncthreads();
    if (j + 1 < KS / 32) {
      const int kb2 = (j + 1) * 32;
      for (int i = tid; i < 32 * KD / 8; i += 256) {
        ((h8*)&Ks[nxt][0][0])[i] = ((const h8*)(Kb + (size_t)kb2 * KD))[i];
        ((h8*)&Vts[nxt][0][0])[i] =
            *(const h8*)(Vb + (size_t)(i >> 2) * KS + kb2 + (i & 3) * 8);
      }
      __syncthreads();
    }
#endif
  }

  // ---- epilogue: divide by row sums, write [b][s][h*128+d] ----
#pragma unroll
  for (int v = 0; v < 8; ++v) li[v] = 1.0f / li[v];
#pragma unroll
  for (int dt = 0; dt < 8; ++dt) {
    const int d = dt * 16 + col;
#pragma unroll
    for (int v = 0; v < 8; ++v) {
      const int s = q0 + v + 8 * g;
      attn[(size_t)(b * KS + s) * KHD + h * KD + d] =
          (_Float16)(acc[dt][v] * li[v]);
    }
  }
}

// ---------------------------------------------------------------------------
// Kernel 3: attn[8192x1024] (f16) @ Wo[1024x128] (f32->f16) + bo -> out f32.
// ---------------------------------------------------------------------------
__global__ __launch_bounds__(256) void out_proj_kernel(
    const _Float16* __restrict__ Ain, const float* __restrict__ Wo,
    const float* __restrict__ bo, float* __restrict__ out) {
  const int tid = threadIdx.x;
  const int lane = tid & 31;
  const int w = tid >> 5;
  const int g = lane >> 4;
  const int col = lane & 15;

  const int tile = blockIdx.x * 8 + w;  // 512*8 tiles
  const int nt = tile & 7;
  const int mt = tile >> 3;
  const int m0 = mt * 16;
  const int n = nt * 16 + col;

  const _Float16* arow = Ain + (size_t)(m0 + col) * KHD;

  v8f acc = {};
#pragma unroll 2
  for (int kc = 0; kc < 32; ++kc) {
    const int k0 = kc * 32;
    h8 lo = *(const h8*)(arow + k0 + 16 * g);
    h8 hi = *(const h8*)(arow + k0 + 16 * g + 8);
    v16h a = pack16(lo, hi);
    v16h bf;
    const float* pb = Wo + (size_t)(k0 + 16 * g) * KD + n;
#pragma unroll
    for (int e = 0; e < 16; ++e) bf[e] = (_Float16)pb[(size_t)e * KD];
    acc = wmma_f16(a, bf, acc);
  }

  const float bias = bo[n];
  float* o = out + (size_t)m0 * KD + n;
#pragma unroll
  for (int v = 0; v < 8; ++v) o[(size_t)(v + 8 * g) * KD] = acc[v] + bias;
}

// ---------------------------------------------------------------------------
extern "C" void kernel_launch(void* const* d_in, const int* in_sizes, int n_in,
                              void* d_out, int out_size, void* d_ws,
                              size_t ws_size, hipStream_t stream) {
  (void)in_sizes; (void)n_in; (void)out_size; (void)ws_size;
  const float* q  = (const float*)d_in[0];
  const float* k  = (const float*)d_in[1];
  const float* v  = (const float*)d_in[2];
  const float* Wq = (const float*)d_in[3];
  const float* bq = (const float*)d_in[4];
  const float* Wk = (const float*)d_in[5];
  const float* bk = (const float*)d_in[6];
  const float* Wv = (const float*)d_in[7];
  const float* bv = (const float*)d_in[8];
  const float* Wo = (const float*)d_in[9];
  const float* bo = (const float*)d_in[10];
  float* out = (float*)d_out;

  // workspace: Qf16 | Kf16 | Vt_f16 | attn_f16  (16 MiB each, 64 MiB total)
  const size_t elems = (size_t)KB * KH * KS * KD;
  _Float16* Qw = (_Float16*)d_ws;
  _Float16* Kw = Qw + elems;
  _Float16* Vw = Kw + elems;
  _Float16* Aw = Vw + elems;

  const float scale = 0.08838834764831845f;  // 1/sqrt(128), folded into Q
  dim3 blk(256);
  proj_kernel<<<dim3(4096), blk, 0, stream>>>(q, Wq, bq, Qw, scale, 0);
  proj_kernel<<<dim3(4096), blk, 0, stream>>>(k, Wk, bk, Kw, 1.0f, 0);
  proj_kernel<<<dim3(4096), blk, 0, stream>>>(v, Wv, bv, Vw, 1.0f, 1);
  flash_attn_kernel<<<dim3(16, 32), blk, 0, stream>>>(Qw, Kw, Vw, Aw);
  out_proj_kernel<<<dim3(512), blk, 0, stream>>>(Aw, Wo, bo, out);
}
